// Attention_61581241090081
// MI455X (gfx1250) — matrix operand order repaired and checked by reading, not compiled, not run on hardware
//
#include <hip/hip_runtime.h>
#include <math.h>

// ---------------- CDNA5 types ----------------
typedef __attribute__((ext_vector_type(16))) __bf16          v16bf;
typedef __attribute__((ext_vector_type(16))) unsigned short  v16u;
typedef __attribute__((ext_vector_type(8)))  float           v8f;
typedef __attribute__((ext_vector_type(4)))  int             v4i;

#define S_LEN   4096
#define DMODEL  2048
#define NHEAD   16
#define DHEAD   128
#define FDIM    2048   // NHEAD * DHEAD
#define BM      128    // GEMM M block
#define KC      32     // GEMM K chunk (= WMMA K)
#define ABN     128    // attention key-block width

#if defined(__has_builtin)
#if __has_builtin(__builtin_amdgcn_global_load_async_to_lds_b128) && \
    __has_builtin(__builtin_amdgcn_s_wait_asynccnt)
#define USE_ASYNC_LDS 1
#endif
#endif
#ifndef USE_ASYNC_LDS
#define USE_ASYNC_LDS 0
#endif

// ---------------- helpers ----------------
__device__ __forceinline__ unsigned short f32_to_bf16(float f) {
  unsigned u = __builtin_bit_cast(unsigned, f);
  u += 0x7FFFu + ((u >> 16) & 1u);          // round-to-nearest-even
  return (unsigned short)(u >> 16);
}

__device__ __forceinline__ v8f wmma_bf16(v16bf a, v16bf b, v8f c) {
  return __builtin_amdgcn_wmma_f32_16x16x32_bf16(
      /*neg_a=*/false, a, /*neg_b=*/false, b,
      /*c_mod=*/(short)0, c, /*reuse_a=*/false, /*reuse_b=*/false);
}

// A fragment (16x32, MxK), source row-major: elem(m,k) = src[m*ld + k]
// ISA 7.12.2: lanes 0-15 row m=lane hold K 0-7 & 16-23; lanes 16-31 hold K 8-15 & 24-31
__device__ __forceinline__ v16bf frag_a(const unsigned short* src, int ld) {
  int lane = threadIdx.x & 31;
  int m  = lane & 15;
  int kb = (lane & 16) ? 8 : 0;
  v16u a;
#pragma unroll
  for (int t = 0; t < 8; ++t) {
    a[t]     = src[m * ld + kb + t];
    a[t + 8] = src[m * ld + kb + t + 16];
  }
  return __builtin_bit_cast(v16bf, a);
}

// B fragment (32x16, KxN) from N-major storage: elem(k,n) = src[n*ld + k]
// lanes 0-15: n=lane, K 0-7 & 16-23 ; lanes 16-31: n=lane-16, K 8-15 & 24-31
__device__ __forceinline__ v16bf frag_b_nmajor(const unsigned short* src, int ld) {
  int lane = threadIdx.x & 31;
  int n  = lane & 15;
  int kb = (lane & 16) ? 8 : 0;
  v16u b;
#pragma unroll
  for (int t = 0; t < 8; ++t) {
    b[t]     = src[n * ld + kb + t];
    b[t + 8] = src[n * ld + kb + t + 16];
  }
  return __builtin_bit_cast(v16bf, b);
}

// ---------------- fp32 -> bf16 conversion ----------------
__global__ __launch_bounds__(256) void cvt_f32_bf16(const float* __restrict__ src,
                                                    unsigned short* __restrict__ dst,
                                                    int n) {
  int i = (blockIdx.x * blockDim.x + threadIdx.x) * 4;
  if (i + 3 < n) {
    float4 f = *(const float4*)(src + i);
    ushort4 o;
    o.x = f32_to_bf16(f.x); o.y = f32_to_bf16(f.y);
    o.z = f32_to_bf16(f.z); o.w = f32_to_bf16(f.w);
    *(ushort4*)(dst + i) = o;
  }
}

// ---------------- QKV projection GEMM ----------------
// grid = (S/128, NHEAD, 3)  block = 256 (8 waves)
// C[head, m, n] = sum_d x[m,d] * W[head, n, d]   (W is N-major over d)
__global__ __launch_bounds__(256) void qkv_gemm(
    const unsigned short* __restrict__ xb,
    const unsigned short* __restrict__ wq,
    const unsigned short* __restrict__ wk,
    const unsigned short* __restrict__ wv,
    unsigned short* __restrict__ qo,
    unsigned short* __restrict__ ko,
    unsigned short* __restrict__ vo) {
  __shared__ __align__(16) unsigned short lA[BM * KC];
  __shared__ __align__(16) unsigned short lB[DHEAD * KC];

  int head = blockIdx.y;
  int m0   = blockIdx.x * BM;
  const unsigned short* w =
      (blockIdx.z == 0 ? wq : (blockIdx.z == 1 ? wk : wv)) + (size_t)head * DHEAD * DMODEL;
  unsigned short* out = (blockIdx.z == 0 ? qo : (blockIdx.z == 1 ? ko : vo));

  int tid  = threadIdx.x;
  int wave = tid >> 5;
  int lane = tid & 31;

  v8f acc[8] = {};

  int row = tid >> 1;
  int c   = (tid & 1) * 16;

  for (int k0 = 0; k0 < DMODEL; k0 += KC) {
    // stage A (128x32) and B (128x32, N-major) tiles
    {
      const uint4* ga = (const uint4*)(xb + (size_t)(m0 + row) * DMODEL + k0 + c);
      uint4* da = (uint4*)(lA + row * KC + c);
      da[0] = ga[0]; da[1] = ga[1];
      const uint4* gb = (const uint4*)(w + (size_t)row * DMODEL + k0 + c);
      uint4* db = (uint4*)(lB + row * KC + c);
      db[0] = gb[0]; db[1] = gb[1];
    }
    if (k0 + KC < DMODEL) {
      __builtin_prefetch(xb + (size_t)(m0 + row) * DMODEL + k0 + KC, 0, 0);
      __builtin_prefetch(w + (size_t)row * DMODEL + k0 + KC, 0, 0);
    }
    __syncthreads();

    // batch-load A + all 8 B fragments (distinct live ranges => loads stay in
    // flight), then burst 8 independent WMMAs
    v16bf a = frag_a(lA + wave * 16 * KC, KC);
    v16bf bfr[8];
#pragma unroll
    for (int nt = 0; nt < 8; ++nt) bfr[nt] = frag_b_nmajor(lB + nt * 16 * KC, KC);
#pragma unroll
    for (int nt = 0; nt < 8; ++nt) acc[nt] = wmma_bf16(a, bfr[nt], acc[nt]);
    __syncthreads();
  }

  int n_lane = lane & 15;
  int rbase  = (lane & 16) ? 8 : 0;
#pragma unroll
  for (int nt = 0; nt < 8; ++nt)
#pragma unroll
    for (int r = 0; r < 8; ++r) {
      int rr = m0 + wave * 16 + rbase + r;
      out[((size_t)head * S_LEN + rr) * DHEAD + nt * 16 + n_lane] = f32_to_bf16(acc[nt][r]);
    }
}

// ---------------- causal flash attention ----------------
// grid = (S/128, NHEAD)  block = 256 (8 waves; each wave owns 16 query rows)
__global__ __launch_bounds__(256) void attn_kernel(
    const unsigned short* __restrict__ q,   // (H, S, DH) bf16
    const unsigned short* __restrict__ kk,  // (H, S, DH) bf16
    const unsigned short* __restrict__ vv,  // (H, S, DH) bf16
    unsigned short* __restrict__ z) {       // (S, FDIM) bf16
  __shared__ __align__(16) unsigned short lK[ABN * DHEAD];   // 32KB; P overlaid after scores
  __shared__ __align__(16) unsigned short lVt[DHEAD * ABN];  // 32KB; V TRANSPOSED: lVt[dh][p]

  int head = blockIdx.y;
  int q0   = blockIdx.x * 128;
  int tid  = threadIdx.x, wave = tid >> 5, lane = tid & 31;
  int n_lane = lane & 15;
  int rbase  = (lane & 16) ? 8 : 0;

  const unsigned short* qh = q  + (size_t)head * S_LEN * DHEAD;
  const unsigned short* kh = kk + (size_t)head * S_LEN * DHEAD;
  const unsigned short* vh = vv + (size_t)head * S_LEN * DHEAD;

  // keep Q fragments (16 rows x 128) resident
  v16bf qf[4];
#pragma unroll
  for (int ch = 0; ch < 4; ++ch)
    qf[ch] = frag_a(qh + (size_t)(q0 + wave * 16) * DHEAD + ch * 32, DHEAD);

  float m_run[8], l_run[8];
  v8f zacc[8] = {};
#pragma unroll
  for (int r = 0; r < 8; ++r) { m_run[r] = -3.0e38f; l_run[r] = 0.f; }

  const float scale = 0.08838834764831845f;  // 1/sqrt(128)
  int kb_max = blockIdx.x;                   // causal: only key blocks <= query block

  for (int kb = 0; kb <= kb_max; ++kb) {
    int kpos0 = kb * ABN;

    // ---- stage K block (async global->LDS where available) ----
    {
      int row = tid >> 1;
      int c   = (tid & 1) * 64;
      const unsigned short* gk = kh + (size_t)(kpos0 + row) * DHEAD + c;
      unsigned short* dk = lK + row * DHEAD + c;
#if USE_ASYNC_LDS
#pragma unroll
      for (int j = 0; j < 8; ++j)
        __builtin_amdgcn_global_load_async_to_lds_b128(
            (__attribute__((address_space(1))) v4i*)(gk + j * 8),
            (__attribute__((address_space(3))) v4i*)(dk + j * 8), 0, 0);
#else
#pragma unroll
      for (int j = 0; j < 8; ++j) ((uint4*)dk)[j] = ((const uint4*)gk)[j];
#endif
    }
    // ---- stage V block TRANSPOSED: lVt[dh * ABN + p]  (register 2x32 transpose) ----
    {
      int pr = (tid & 63) * 2;   // row pair 0..126
      int c0 = (tid >> 6) * 32;  // dh group: 0,32,64,96
      const uint4* g0 = (const uint4*)(vh + (size_t)(kpos0 + pr) * DHEAD + c0);
      const uint4* g1 = (const uint4*)(vh + (size_t)(kpos0 + pr + 1) * DHEAD + c0);
#pragma unroll
      for (int j = 0; j < 4; ++j) {
        uint4 a0 = g0[j], a1 = g1[j];
        const unsigned u0[4] = {a0.x, a0.y, a0.z, a0.w};
        const unsigned u1[4] = {a1.x, a1.y, a1.z, a1.w};
#pragma unroll
        for (int e = 0; e < 4; ++e) {
          int d = c0 + j * 8 + e * 2;
          unsigned lo = (u0[e] & 0xFFFFu) | (u1[e] << 16);
          unsigned hi = (u0[e] >> 16) | (u1[e] & 0xFFFF0000u);
          *(unsigned*)(lVt + (size_t)d * ABN + pr)       = lo;
          *(unsigned*)(lVt + (size_t)(d + 1) * ABN + pr) = hi;
        }
      }
    }
#if USE_ASYNC_LDS
    __builtin_amdgcn_s_wait_asynccnt(0);
#endif
    __syncthreads();

    // ---- scores S = Q . K^T  (16 x 128 per wave); batch loads + WMMA burst ----
    v8f sc[8] = {};
#pragma unroll
    for (int ch = 0; ch < 4; ++ch) {
      v16bf bfr[8];
#pragma unroll
      for (int nt = 0; nt < 8; ++nt)
        bfr[nt] = frag_b_nmajor(lK + nt * 16 * DHEAD + ch * 32, DHEAD);
#pragma unroll
      for (int nt = 0; nt < 8; ++nt) sc[nt] = wmma_bf16(qf[ch], bfr[nt], sc[nt]);
    }
    __syncthreads();  // everyone done reading lK before P overlays it

    // ---- scale, causal mask, online softmax ----
    bool diag = (kb == kb_max);
#pragma unroll
    for (int r = 0; r < 8; ++r) {
      int qrow = q0 + wave * 16 + rbase + r;
      float mx = -3.0e38f;
#pragma unroll
      for (int nt = 0; nt < 8; ++nt) {
        float s = sc[nt][r] * scale;
        if (diag && (kpos0 + nt * 16 + n_lane) > qrow) s = -3.0e38f;
        sc[nt][r] = s;
        mx = fmaxf(mx, s);
      }
      mx = fmaxf(mx, __shfl_xor(mx, 1, 32));
      mx = fmaxf(mx, __shfl_xor(mx, 2, 32));
      mx = fmaxf(mx, __shfl_xor(mx, 4, 32));
      mx = fmaxf(mx, __shfl_xor(mx, 8, 32));
      float mnew = fmaxf(m_run[r], mx);
      float corr = __expf(m_run[r] - mnew);
      float rowsum = 0.f;
#pragma unroll
      for (int nt = 0; nt < 8; ++nt) {
        float p = __expf(sc[nt][r] - mnew);
        sc[nt][r] = p;
        rowsum += p;
      }
      rowsum += __shfl_xor(rowsum, 1, 32);
      rowsum += __shfl_xor(rowsum, 2, 32);
      rowsum += __shfl_xor(rowsum, 4, 32);
      rowsum += __shfl_xor(rowsum, 8, 32);
      l_run[r] = l_run[r] * corr + rowsum;
      m_run[r] = mnew;
#pragma unroll
      for (int nt = 0; nt < 8; ++nt) zacc[nt][r] *= corr;
    }

    // ---- P -> LDS (each wave's private 16x128 region, overlaid on lK) ----
    unsigned short* pReg = lK + wave * (16 * ABN);
#pragma unroll
    for (int nt = 0; nt < 8; ++nt)
#pragma unroll
      for (int r = 0; r < 8; ++r)
        pReg[(rbase + r) * ABN + nt * 16 + n_lane] = f32_to_bf16(sc[nt][r]);

    // ---- Z += P . V   (V transposed in LDS => contiguous B fragments) ----
#pragma unroll
    for (int ch = 0; ch < 4; ++ch) {
      v16bf pa = frag_a(pReg + ch * 32, ABN);
      v16bf bfr[8];
#pragma unroll
      for (int nt = 0; nt < 8; ++nt)
        bfr[nt] = frag_b_nmajor(lVt + nt * 16 * ABN + ch * 32, ABN);
#pragma unroll
      for (int nt = 0; nt < 8; ++nt) zacc[nt] = wmma_bf16(pa, bfr[nt], zacc[nt]);
    }
    __syncthreads();  // done with lVt / P region before next staging
  }

  // ---- normalize and store z (S, FDIM) with head offset ----
#pragma unroll
  for (int nt = 0; nt < 8; ++nt)
#pragma unroll
    for (int r = 0; r < 8; ++r) {
      int qrow = q0 + wave * 16 + rbase + r;
      float val = zacc[nt][r] / l_run[r];
      z[(size_t)qrow * FDIM + head * DHEAD + nt * 16 + n_lane] = f32_to_bf16(val);
    }
}

// ---------------- output projection GEMM ----------------
// grid = (S/128, DMODEL/128)  block = 256
// out[m, n0+n] = sum_f z[m,f] * WO[n0+n, f]   (WO N-major over f), fp32 out
__global__ __launch_bounds__(256) void oproj_gemm(
    const unsigned short* __restrict__ zb,
    const unsigned short* __restrict__ wob,
    float* __restrict__ out) {
  __shared__ __align__(16) unsigned short lA[BM * KC];
  __shared__ __align__(16) unsigned short lB[128 * KC];

  int m0 = blockIdx.x * BM;
  int n0 = blockIdx.y * 128;
  const unsigned short* w = wob + (size_t)n0 * FDIM;

  int tid = threadIdx.x, wave = tid >> 5, lane = tid & 31;
  v8f acc[8] = {};
  int row = tid >> 1;
  int c   = (tid & 1) * 16;

  for (int k0 = 0; k0 < FDIM; k0 += KC) {
    {
      const uint4* ga = (const uint4*)(zb + (size_t)(m0 + row) * FDIM + k0 + c);
      uint4* da = (uint4*)(lA + row * KC + c);
      da[0] = ga[0]; da[1] = ga[1];
      const uint4* gb = (const uint4*)(w + (size_t)row * FDIM + k0 + c);
      uint4* db = (uint4*)(lB + row * KC + c);
      db[0] = gb[0]; db[1] = gb[1];
    }
    if (k0 + KC < FDIM) {
      __builtin_prefetch(zb + (size_t)(m0 + row) * FDIM + k0 + KC, 0, 0);
      __builtin_prefetch(w + (size_t)row * FDIM + k0 + KC, 0, 0);
    }
    __syncthreads();

    v16bf a = frag_a(lA + wave * 16 * KC, KC);
    v16bf bfr[8];
#pragma unroll
    for (int nt = 0; nt < 8; ++nt) bfr[nt] = frag_b_nmajor(lB + nt * 16 * KC, KC);
#pragma unroll
    for (int nt = 0; nt < 8; ++nt) acc[nt] = wmma_bf16(a, bfr[nt], acc[nt]);
    __syncthreads();
  }

  int n_lane = lane & 15;
  int rbase  = (lane & 16) ? 8 : 0;
#pragma unroll
  for (int nt = 0; nt < 8; ++nt)
#pragma unroll
    for (int r = 0; r < 8; ++r) {
      int rr = m0 + wave * 16 + rbase + r;
      out[(size_t)rr * DMODEL + n0 + nt * 16 + n_lane] = acc[nt][r];
    }
}

// ---------------- host launch ----------------
extern "C" void kernel_launch(void* const* d_in, const int* in_sizes, int n_in,
                              void* d_out, int out_size, void* d_ws, size_t ws_size,
                              hipStream_t stream) {
  // setup_inputs order: x, W_K, W_Q, W_V, W_O
  const float* x  = (const float*)d_in[0];
  const float* WK = (const float*)d_in[1];
  const float* WQ = (const float*)d_in[2];
  const float* WV = (const float*)d_in[3];
  const float* WO = (const float*)d_in[4];
  float* out = (float*)d_out;

  // workspace layout (bf16 staging, 112MB total — intermediates live in the 192MB L2)
  char* ws = (char*)d_ws;
  unsigned short* xb  = (unsigned short*)(ws);                           // 16MB
  unsigned short* wqb = (unsigned short*)(ws + (size_t)16 * (1 << 20));  // 8MB
  unsigned short* wkb = (unsigned short*)(ws + (size_t)24 * (1 << 20));  // 8MB
  unsigned short* wvb = (unsigned short*)(ws + (size_t)32 * (1 << 20));  // 8MB
  unsigned short* wob = (unsigned short*)(ws + (size_t)40 * (1 << 20));  // 8MB
  unsigned short* qb  = (unsigned short*)(ws + (size_t)48 * (1 << 20));  // 16MB
  unsigned short* kb  = (unsigned short*)(ws + (size_t)64 * (1 << 20));  // 16MB
  unsigned short* vb  = (unsigned short*)(ws + (size_t)80 * (1 << 20));  // 16MB
  unsigned short* zb  = (unsigned short*)(ws + (size_t)96 * (1 << 20));  // 16MB

  const int n_x = S_LEN * DMODEL;           // 8M
  const int n_w = NHEAD * DHEAD * DMODEL;   // 4M
  const int n_o = DMODEL * FDIM;            // 4M

  cvt_f32_bf16<<<dim3((n_x / 4 + 255) / 256), dim3(256), 0, stream>>>(x, xb, n_x);
  cvt_f32_bf16<<<dim3((n_w / 4 + 255) / 256), dim3(256), 0, stream>>>(WQ, wqb, n_w);
  cvt_f32_bf16<<<dim3((n_w / 4 + 255) / 256), dim3(256), 0, stream>>>(WK, wkb, n_w);
  cvt_f32_bf16<<<dim3((n_w / 4 + 255) / 256), dim3(256), 0, stream>>>(WV, wvb, n_w);
  cvt_f32_bf16<<<dim3((n_o / 4 + 255) / 256), dim3(256), 0, stream>>>(WO, wob, n_o);

  qkv_gemm<<<dim3(S_LEN / BM, NHEAD, 3), dim3(256), 0, stream>>>(
      xb, wqb, wkb, wvb, qb, kb, vb);

  attn_kernel<<<dim3(S_LEN / 128, NHEAD), dim3(256), 0, stream>>>(qb, kb, vb, zb);

  oproj_gemm<<<dim3(S_LEN / BM, DMODEL / 128), dim3(256), 0, stream>>>(zb, wob, out);
}
